// GatEncoder_32238024524374
// MI455X (gfx1250) — matrix operand order repaired and checked by reading, not compiled, not run on hardware
//
#include <hip/hip_runtime.h>

// ---------------- problem constants (from reference) ----------------
#define N_NODES 5000
#define N_EDGES 15000
#define FDIM_IN 300
#define GCH     2048
#define H_HEADS 4
#define HID     2048
#define NOUT    768
#define NGR     128
#define HGCH    (H_HEADS * GCH)        // 8192
#define EPLUS   (N_EDGES + N_NODES)    // edges incl. self loops = 20000
#define NEG_SLOPE 0.2f

typedef __attribute__((ext_vector_type(16))) __bf16 v16bf;
typedef __attribute__((ext_vector_type(4)))  __bf16 v4bf;
typedef __attribute__((ext_vector_type(8)))  float  v8f;

// =====================================================================
// Pack A (fp32 row-major [M,K]) -> bf16 WMMA A-fragment tile order:
//   Ap[ ((row*KT + kt)*2 + h)*16 + e ]
// where for k = kt*32 + kk:  h = kk>>4, e = kk&15
// A lane (lh = lane>>4) then reads its 16-element fragment as ONE contiguous
// 32-byte chunk. Rows >= M and k >= K are zero-filled here so the GEMM inner
// loop is guard-free.
// =====================================================================
__launch_bounds__(256)
__global__ void pack_a_bf16(const float* __restrict__ A, __bf16* __restrict__ Ap,
                            int M, int Mp, int K, int KT)
{
    long idx = (long)blockIdx.x * 256 + threadIdx.x;     // one k-quad per thread
    long qpr = (long)KT * 8;                             // quads per row
    if (idx >= (long)Mp * qpr) return;
    int row = (int)(idx / qpr);
    int quad = (int)(idx % qpr);
    int kt = quad >> 3;
    int kk = (quad & 7) << 2;                            // k within tile, 0..28 step 4
    int gk = (kt << 5) + kk;

    float x0 = 0.f, x1 = 0.f, x2 = 0.f, x3 = 0.f;
    if (row < M) {
        const float* ap = A + (size_t)row * K + gk;
        if (gk + 3 < K) { float4 v = *(const float4*)ap; x0 = v.x; x1 = v.y; x2 = v.z; x3 = v.w; }
        else {
            if (gk + 0 < K) x0 = ap[0];
            if (gk + 1 < K) x1 = ap[1];
            if (gk + 2 < K) x2 = ap[2];
            if (gk + 3 < K) x3 = ap[3];
        }
    }
    int q = kk & 15;
    int hh = kk >> 4;
    int e  = kk & 15;                                    // e in {0,4,8,12}: 8B aligned
    v4bf t = { (__bf16)x0, (__bf16)x1, (__bf16)x2, (__bf16)x3 };
    *(v4bf*)&Ap[(((size_t)row * KT + kt) * 2 + hh) * 16 + e] = t;
}

// =====================================================================
// Pack B (fp32 row-major [K,N]) -> bf16 WMMA B-fragment tile order:
//   Bp[ ((nt*KT + kt)*32 + lane)*16 + e ]
// lane = (n&15) + 16*(k>=16), e = k&15. Lane's fragment = ONE contiguous 32B;
// each (nt,kt) chunk = 1KB contiguous (async-copy friendly).
// =====================================================================
__launch_bounds__(256)
__global__ void pack_b_bf16(const float* __restrict__ B, __bf16* __restrict__ Bp,
                            int K, int KT, int Nn)
{
    long idx = (long)blockIdx.x * 256 + threadIdx.x;
    if (idx >= (long)KT * 8 * Nn) return;
    int n = (int)(idx % Nn);
    long rest = idx / Nn;
    int kt = (int)(rest >> 3);
    int kk = ((int)rest & 7) << 2;
    int gk = (kt << 5) + kk;

    float x0 = (gk + 0 < K) ? B[(size_t)(gk + 0) * Nn + n] : 0.f;
    float x1 = (gk + 1 < K) ? B[(size_t)(gk + 1) * Nn + n] : 0.f;
    float x2 = (gk + 2 < K) ? B[(size_t)(gk + 2) * Nn + n] : 0.f;
    float x3 = (gk + 3 < K) ? B[(size_t)(gk + 3) * Nn + n] : 0.f;

    int nt   = n >> 4;
    int lane = (n & 15) + ((kk >= 16) ? 16 : 0);
    int e    = kk & 15;                                  // e in {0,4,8,12}
    v4bf t = { (__bf16)x0, (__bf16)x1, (__bf16)x2, (__bf16)x3 };
    *(v4bf*)&Bp[(((size_t)nt * KT + kt) * 32 + lane) * 16 + e] = t;
}

// =====================================================================
// GEMM on packed bf16 fragments: C = act(A @ B + bias).
// Block = 256 threads = 8 waves, block tile 128(M) x 128(N).
// Wave w owns rows [m0+16w, m0+16w+16) and all 8 WMMA column tiles.
// B is staged in LDS (double-buffered) with CDNA5 async global->LDS copies
// (global_load_async_to_lds_b128, tracked by ASYNCcnt) and shared by all 8
// waves; A fragments stream directly from the L2-resident packed buffer as
// contiguous 32B loads. Inner loop: 8 v_wmma per K-step, no guards.
// =====================================================================
__launch_bounds__(256)
__global__ void gemm_packed_wmma(const __bf16* __restrict__ Ap, const __bf16* __restrict__ Bp,
                                 const float* __restrict__ bias, float* __restrict__ C,
                                 int M, int Nn, int KT, int relu)
{
    // double-buffered B stage: 2 x 8 tiles x 512 bf16 (2 x 8KB)
    __shared__ __attribute__((aligned(32))) __bf16 Bs[2 * 8 * 512];

    const int tid  = threadIdx.x;
    const int lane = tid & 31;
    const int wv   = tid >> 5;
    const int lm   = lane & 15;
    const int lh   = lane >> 4;
    const int m0   = blockIdx.x * 128;
    const int n0   = blockIdx.y * 128;
    const int nt0  = n0 >> 4;                            // first of 8 B tiles

    const int r = m0 + (wv << 4) + lm;                   // A fragment row (always < Mp)
    const __bf16* ap = Ap + ((size_t)r * KT * 2 + lh) * 16;

    // staging descriptor for this thread: 2 chunks of 16B out of 8KB per K-step
    const int c0 = tid << 1;
    const int s_nt[2] = { (c0 + 0) >> 6, (c0 + 1) >> 6 };
    const int s_wi[2] = { (c0 + 0) & 63, (c0 + 1) & 63 };

    v8f acc[8] = {};

    // prologue: stage kt=0 into buffer 0
#pragma unroll
    for (int i = 0; i < 2; ++i) {
        const __bf16* src = Bp + (((size_t)(nt0 + s_nt[i]) * KT + 0) * 512 + s_wi[i] * 8);
        unsigned dst = (unsigned)(uintptr_t)&Bs[s_nt[i] * 512 + s_wi[i] * 8];
        asm volatile("global_load_async_to_lds_b128 %0, %1, off" :: "v"(dst), "v"(src) : "memory");
    }
    asm volatile("s_wait_asynccnt 0" ::: "memory");
    __syncthreads();

    for (int kt = 0; kt < KT; ++kt) {
        const int pb = kt & 1;
        // issue async stage of next K-step into the other buffer
        if (kt + 1 < KT) {
#pragma unroll
            for (int i = 0; i < 2; ++i) {
                const __bf16* src = Bp + (((size_t)(nt0 + s_nt[i]) * KT + (kt + 1)) * 512 + s_wi[i] * 8);
                unsigned dst = (unsigned)(uintptr_t)&Bs[(pb ^ 1) * 4096 + s_nt[i] * 512 + s_wi[i] * 8];
                asm volatile("global_load_async_to_lds_b128 %0, %1, off" :: "v"(dst), "v"(src) : "memory");
            }
        }
        if (kt + 8 < KT) __builtin_prefetch(ap + (size_t)(kt + 8) * 32, 0, 1);

        // compute: A direct from global, B fragments from LDS
        v16bf a = *(const v16bf*)(ap + (size_t)kt * 32);
#pragma unroll
        for (int nt = 0; nt < 8; ++nt) {
            v16bf b = *(const v16bf*)&Bs[pb * 4096 + nt * 512 + lane * 16];
            acc[nt] = __builtin_amdgcn_wmma_f32_16x16x32_bf16(
                false, a, false, b, (short)0, acc[nt], false, false);
        }

        // next buffer ready + all waves done reading this one
        asm volatile("s_wait_asynccnt 0" ::: "memory");
        __syncthreads();
    }

    // epilogue: C/D layout -> VGPR v holds M = v + 8*(lane>>4), N = lane&15
#pragma unroll
    for (int nt = 0; nt < 8; ++nt) {
        int c = n0 + (nt << 4) + lm;
        float bv = (bias != nullptr) ? bias[c] : 0.0f;
#pragma unroll
        for (int v = 0; v < 8; ++v) {
            int rr = m0 + (wv << 4) + v + (lh << 3);
            if (rr < M) {
                float o = acc[nt][v] + bv;
                if (relu) o = fmaxf(o, 0.0f);
                C[(size_t)rr * Nn + c] = o;
            }
        }
    }
}

// ---------------- attention scores: a_src/a_dst[n,h] = sum_c h[n,h,c]*att[h,c] ----------------
__launch_bounds__(256)
__global__ void attn_scores(const float* __restrict__ hfeat,
                            const float* __restrict__ att_src,
                            const float* __restrict__ att_dst,
                            float* __restrict__ asrc, float* __restrict__ adst)
{
    __shared__ float r1[256], r2[256];
    const int node = blockIdx.x;
    const int head = blockIdx.y;
    const float* hp = hfeat + (size_t)node * HGCH + (size_t)head * GCH;
    const float* sp = att_src + head * GCH;
    const float* dp = att_dst + head * GCH;
    float s1 = 0.0f, s2 = 0.0f;
    for (int c = threadIdx.x; c < GCH; c += 256) {
        float v = hp[c];
        s1 += v * sp[c];
        s2 += v * dp[c];
    }
    r1[threadIdx.x] = s1; r2[threadIdx.x] = s2;
    __syncthreads();
    for (int st = 128; st > 0; st >>= 1) {
        if (threadIdx.x < st) {
            r1[threadIdx.x] += r1[threadIdx.x + st];
            r2[threadIdx.x] += r2[threadIdx.x + st];
        }
        __syncthreads();
    }
    if (threadIdx.x == 0) {
        asrc[node * H_HEADS + head] = r1[0];
        adst[node * H_HEADS + head] = r2[0];
    }
}

// ---------------- small utility kernels ----------------
__global__ void fill_f32(float* __restrict__ p, float v, long n)
{
    long i = (long)blockIdx.x * 256 + threadIdx.x;
    if (i < n) p[i] = v;
}

__global__ void bcast_bias(float* __restrict__ out, const float* __restrict__ bias, long n, int c)
{
    long i = (long)blockIdx.x * 256 + threadIdx.x;
    if (i < n) out[i] = bias[i % c];
}

__device__ __forceinline__ void atomicMaxF(float* addr, float val)
{
    // orderable-int float max (init must be -inf)
    int iv = __float_as_int(val);
    if (iv >= 0) atomicMax((int*)addr, iv);
    else         atomicMin((unsigned int*)addr, (unsigned int)iv);
}

// ---------------- edge kernels (edge set = edges + self loops) ----------------
__global__ void edge_logit_max(const int* __restrict__ ei,
                               const float* __restrict__ asrc, const float* __restrict__ adst,
                               float* __restrict__ elog, float* __restrict__ mx)
{
    int t = blockIdx.x * 256 + threadIdx.x;
    if (t >= EPLUS * H_HEADS) return;
    int e = t >> 2, h = t & 3;     // H_HEADS == 4
    int s, d;
    if (e < N_EDGES) { s = ei[e]; d = ei[N_EDGES + e]; }
    else             { s = d = e - N_EDGES; }
    float v = asrc[s * H_HEADS + h] + adst[d * H_HEADS + h];
    v = (v > 0.0f) ? v : NEG_SLOPE * v;
    elog[t] = v;
    atomicMaxF(&mx[d * H_HEADS + h], v);
}

__global__ void edge_exp_sum(const int* __restrict__ ei, float* __restrict__ elog,
                             const float* __restrict__ mx, float* __restrict__ ssum)
{
    int t = blockIdx.x * 256 + threadIdx.x;
    if (t >= EPLUS * H_HEADS) return;
    int e = t >> 2, h = t & 3;
    int d = (e < N_EDGES) ? ei[N_EDGES + e] : (e - N_EDGES);
    float ex = __expf(elog[t] - mx[d * H_HEADS + h]);
    elog[t] = ex;
    atomicAdd(&ssum[d * H_HEADS + h], ex);
}

__launch_bounds__(256)
__global__ void scatter_messages(const int* __restrict__ ei, const float* __restrict__ hfeat,
                                 const float* __restrict__ elog, const float* __restrict__ ssum,
                                 float* __restrict__ out)
{
    const int e = blockIdx.x;
    int s, d;
    if (e < N_EDGES) { s = ei[e]; d = ei[N_EDGES + e]; }
    else             { s = d = e - N_EDGES; }
    __shared__ float alpha[H_HEADS];
    if (threadIdx.x < H_HEADS)
        alpha[threadIdx.x] = elog[e * H_HEADS + threadIdx.x] /
                             (ssum[d * H_HEADS + threadIdx.x] + 1e-16f);
    __syncthreads();
    const float* hp = hfeat + (size_t)s * HGCH;
    float* op = out + (size_t)d * HGCH;
    for (int j = threadIdx.x; j < HGCH; j += 256)
        atomicAdd(&op[j], hp[j] * alpha[j >> 11]);   // GCH == 2048 == 2^11
}

// ---------------- pooling & layer norm ----------------
__global__ void pool_sum(const float* __restrict__ hin, const int* __restrict__ batch,
                         float* __restrict__ g, float* __restrict__ cnt)
{
    const int n = blockIdx.x;
    const int b = batch[n];
    const float* hp = hin + (size_t)n * HID;
    float* gp = g + (size_t)b * HID;
    for (int j = threadIdx.x; j < HID; j += 256) atomicAdd(&gp[j], hp[j]);
    if (threadIdx.x == 0) atomicAdd(&cnt[b], 1.0f);
}

__global__ void pool_div(float* __restrict__ g, const float* __restrict__ cnt)
{
    long i = (long)blockIdx.x * 256 + threadIdx.x;
    if (i < (long)NGR * HID) g[i] /= fmaxf(cnt[i / HID], 1.0f);
}

__launch_bounds__(256)
__global__ void layer_norm(const float* __restrict__ x, const float* __restrict__ gm,
                           const float* __restrict__ bt, float* __restrict__ out)
{
    __shared__ float r1[256], r2[256];
    const int row = blockIdx.x;
    const float* xp = x + (size_t)row * NOUT;
    float s = 0.0f, s2 = 0.0f;
    for (int j = threadIdx.x; j < NOUT; j += 256) { float v = xp[j]; s += v; s2 += v * v; }
    r1[threadIdx.x] = s; r2[threadIdx.x] = s2;
    __syncthreads();
    for (int st = 128; st > 0; st >>= 1) {
        if (threadIdx.x < st) { r1[threadIdx.x] += r1[threadIdx.x + st]; r2[threadIdx.x] += r2[threadIdx.x + st]; }
        __syncthreads();
    }
    float mu  = r1[0] / (float)NOUT;
    float var = r2[0] / (float)NOUT - mu * mu;
    float inv = rsqrtf(var + 1e-5f);
    for (int j = threadIdx.x; j < NOUT; j += 256)
        out[(size_t)row * NOUT + j] = (xp[j] - mu) * inv * gm[j] + bt[j];
}

// ---------------- host orchestration ----------------
extern "C" void kernel_launch(void* const* d_in, const int* in_sizes, int n_in,
                              void* d_out, int out_size, void* d_ws, size_t ws_size,
                              hipStream_t stream)
{
    (void)in_sizes; (void)n_in; (void)out_size; (void)ws_size;

    const float* x     = (const float*)d_in[0];
    const int*   ei    = (const int*)d_in[1];
    const int*   batch = (const int*)d_in[2];
    const float* W[3]     = {(const float*)d_in[3],  (const float*)d_in[9],  (const float*)d_in[15]};
    const float* asrcW[3] = {(const float*)d_in[4],  (const float*)d_in[10], (const float*)d_in[16]};
    const float* adstW[3] = {(const float*)d_in[5],  (const float*)d_in[11], (const float*)d_in[17]};
    const float* bc[3]    = {(const float*)d_in[6],  (const float*)d_in[12], (const float*)d_in[18]};
    const float* Wh[3]    = {(const float*)d_in[7],  (const float*)d_in[13], (const float*)d_in[19]};
    const float* bh[3]    = {(const float*)d_in[8],  (const float*)d_in[14], (const float*)d_in[20]};
    const float* Wm1  = (const float*)d_in[21];
    const float* bm1  = (const float*)d_in[22];
    const float* Wm2  = (const float*)d_in[23];
    const float* bm2  = (const float*)d_in[24];
    const float* ln_g = (const float*)d_in[25];
    const float* ln_b = (const float*)d_in[26];
    float* out = (float*)d_out;

    // workspace carve-up
    float* ws = (float*)d_ws;
    size_t o = 0;
    float* hbuf = ws + o; o += (size_t)N_NODES * HGCH;     // x@W result  [N,8192]
    float* gout = ws + o; o += (size_t)N_NODES * HGCH;     // GAT output  [N,8192]
    float* hid  = ws + o; o += (size_t)N_NODES * HID;      // head-linear [N,2048]
    float* asc  = ws + o; o += (size_t)N_NODES * H_HEADS;
    float* adc  = ws + o; o += (size_t)N_NODES * H_HEADS;
    float* mx   = ws + o; o += (size_t)N_NODES * H_HEADS;
    float* ssum = ws + o; o += (size_t)N_NODES * H_HEADS;
    float* elog = ws + o; o += (size_t)EPLUS * H_HEADS;
    float* g    = ws + o; o += (size_t)NGR * HID;
    float* cnt  = ws + o; o += (size_t)NGR;
    float* mlp1 = ws + o; o += (size_t)NGR * HID;
    float* mlp2 = ws + o; o += (size_t)NGR * NOUT;
    // bf16 packed operand buffers (reused per GEMM)
    __bf16* Apk = (__bf16*)(ws + o); o += ((size_t)5120 * 8192) / 2;          // 5120 rows x 8192 bf16
    __bf16* Bpk = (__bf16*)(ws + o); o += ((size_t)8192 * 2048) / 2;          // max 16.8M bf16

    dim3 blk(256);
    auto gemm = [&](const float* A, const float* B, const float* bias, float* C,
                    int M, int Nn, int K, int relu) {
        int KT = (K + 31) / 32;
        int Mp = ((M + 127) / 128) * 128;
        long na = (long)Mp * KT * 8;                       // A pack threads (k-quads)
        long nb = (long)KT * 8 * Nn;                       // B pack threads
        pack_a_bf16<<<(unsigned)((na + 255) / 256), blk, 0, stream>>>(A, Apk, M, Mp, K, KT);
        pack_b_bf16<<<(unsigned)((nb + 255) / 256), blk, 0, stream>>>(B, Bpk, K, KT, Nn);
        dim3 grid(Mp / 128, Nn / 128);
        gemm_packed_wmma<<<grid, blk, 0, stream>>>(Apk, Bpk, bias, C, M, Nn, KT, relu);
    };

    const long nh  = (long)N_NODES * H_HEADS;
    const long eh  = (long)EPLUS * H_HEADS;
    const long tot = (long)N_NODES * HGCH;

    const float* lin = x;
    int lfin = FDIM_IN;
    for (int l = 0; l < 3; ++l) {
        // h = lin @ W_l                         [N, 8192]
        gemm(lin, W[l], nullptr, hbuf, N_NODES, HGCH, lfin, 0);
        // per-(node,head) attention scores
        attn_scores<<<dim3(N_NODES, H_HEADS), blk, 0, stream>>>(hbuf, asrcW[l], adstW[l], asc, adc);
        // segment softmax state init; gout starts at bias (segment_sum + bias)
        fill_f32<<<(unsigned)((nh + 255) / 256), blk, 0, stream>>>(mx, -__builtin_inff(), nh);
        fill_f32<<<(unsigned)((nh + 255) / 256), blk, 0, stream>>>(ssum, 0.0f, nh);
        bcast_bias<<<(unsigned)((tot + 255) / 256), blk, 0, stream>>>(gout, bc[l], tot, HGCH);
        // edge softmax + message scatter
        edge_logit_max<<<(unsigned)((eh + 255) / 256), blk, 0, stream>>>(ei, asc, adc, elog, mx);
        edge_exp_sum<<<(unsigned)((eh + 255) / 256), blk, 0, stream>>>(ei, elog, mx, ssum);
        scatter_messages<<<EPLUS, blk, 0, stream>>>(ei, hbuf, elog, ssum, gout);
        // head linear (+ReLU for layers 1,2)
        gemm(gout, Wh[l], bh[l], hid, N_NODES, HID, HGCH, (l < 2) ? 1 : 0);
        lin = hid; lfin = HID;
    }

    // global mean pool over graphs
    fill_f32<<<(unsigned)(((long)NGR * HID + 255) / 256), blk, 0, stream>>>(g, 0.0f, (long)NGR * HID);
    fill_f32<<<1, blk, 0, stream>>>(cnt, 0.0f, (long)NGR);
    pool_sum<<<N_NODES, blk, 0, stream>>>(hid, batch, g, cnt);
    pool_div<<<(unsigned)(((long)NGR * HID + 255) / 256), blk, 0, stream>>>(g, cnt);

    // MLP head + layer norm
    gemm(g,    Wm1, bm1, mlp1, NGR, HID,  HID, 1);
    gemm(mlp1, Wm2, bm2, mlp2, NGR, NOUT, HID, 0);
    layer_norm<<<NGR, blk, 0, stream>>>(mlp2, ln_g, ln_b, out);
}
